// Encoder_43885975830690
// MI455X (gfx1250) — compile-verified, matrix operand order checked by reading, not run on hardware
//
#include <hip/hip_runtime.h>
#include <hip/hip_bf16.h>

typedef __bf16 bf16_t;
typedef __attribute__((ext_vector_type(16))) __bf16 v16bf;
typedef __attribute__((ext_vector_type(8)))  __bf16 v8bf;
typedef __attribute__((ext_vector_type(8)))  float  v8f;

#define B_   32
#define T_   512
#define D_   1024
#define H_   1024
#define KK_  2048   // D + H
#define NG_  4096   // 4 * H
#define KCH_ 256    // k-chunk staged in LDS by the TDM per recurrent step

// ---- TDM availability -------------------------------------------------------
#define USE_TDM 1
#if defined(__has_builtin)
#if !__has_builtin(__builtin_amdgcn_tensor_load_to_lds)
#undef USE_TDM
#define USE_TDM 0
#endif
#else
#undef USE_TDM
#define USE_TDM 0
#endif

#if USE_TDM
typedef unsigned int v4u __attribute__((ext_vector_type(4)));
typedef int          v8i __attribute__((ext_vector_type(8)));
typedef int          v4i __attribute__((ext_vector_type(4)));

// Issue one 2D TDM tile load: rows x cols (bf16 elements), row stride in elements.
// D# per CDNA5 ISA 8.3/8.4: group0 = {count, lds_addr, global_addr, type=2},
// group1 = {data_size=2B, tensor_dim0/1, tile_dim0/1, tensor_dim0_stride}.
// This toolchain's builtin takes 6 args (g0, g1, g2, g3, g4, cpol); the upper
// descriptor groups are unused for a 2D tile -> zero-filled.
__device__ __forceinline__ void tdm_load_tile(unsigned lds_off, const bf16_t* gsrc,
                                              int rows, int cols, int stride) {
    unsigned long long ga = (unsigned long long)(uintptr_t)gsrc;
    v4u g0;
    g0.x = 1u;                                                  // count=1 (valid user D#)
    g0.y = lds_off;                                             // LDS byte address
    g0.z = (unsigned)ga;                                        // global_addr[31:0]
    g0.w = ((unsigned)(ga >> 32) & 0x01FFFFFFu) | (2u << 30);   // addr[56:32] | type=2
    v8i g1;
    g1[0] = 0x00010000;                                         // data_size=1 -> 2 bytes
    g1[1] = (int)((cols & 0xFFFF) << 16);                       // tensor_dim0 lo
    g1[2] = (int)(((cols >> 16) & 0xFFFF) | ((rows & 0xFFFF) << 16)); // dim0 hi | dim1 lo
    g1[3] = (int)(((rows >> 16) & 0xFFFF) | ((cols & 0xFFFF) << 16)); // dim1 hi | tile_dim0
    g1[4] = rows & 0xFFFF;                                      // tile_dim1 (tile_dim2=0)
    g1[5] = stride;                                             // tensor_dim0_stride[31:0]
    g1[6] = 0;
    g1[7] = 0;
    v4i gz4 = {0, 0, 0, 0};
    v8i gz8 = {0, 0, 0, 0, 0, 0, 0, 0};
    __builtin_amdgcn_tensor_load_to_lds(g0, g1, gz4, gz4, gz8, 0);
}
#endif

// Per-lane bf16 A/B fragment for v_wmma_f32_16x16x32_bf16 (wave32):
// lanes 0-15 hold K = {0..7, 16..23}, lanes 16-31 hold K = {8..15, 24..31}.
// Caller passes p = row_base + k_tile + ((lane>>4)<<3).
__device__ __forceinline__ v16bf load_frag(const bf16_t* p) {
    v8bf lo = *(const v8bf*)(p);
    v8bf hi = *(const v8bf*)(p + 16);
    v16bf r;
#pragma unroll
    for (int i = 0; i < 8; ++i) { r[i] = lo[i]; r[i + 8] = hi[i]; }
    return r;
}

__device__ __forceinline__ float sigm(float x) {
    return 1.0f / (1.0f + __expf(-x));
}

// ---- prep: fp32 -> bf16 copy of X ------------------------------------------
__global__ void k_cvt_x(const float* __restrict__ x, bf16_t* __restrict__ xb, int n) {
    int i = blockIdx.x * blockDim.x + threadIdx.x;
    if (i < n) xb[i] = (bf16_t)x[i];
}

// ---- prep: transpose+convert kernel [KK,4H] f32 -> Wt [4H,KK] bf16 ----------
__global__ void k_transpose_w(const float* __restrict__ w, bf16_t* __restrict__ wt) {
    int i = blockIdx.x * blockDim.x + threadIdx.x;   // i = n*KK_ + k, exact grid
    int n = i / KK_;
    int k = i - n * KK_;
    wt[i] = (bf16_t)w[(size_t)k * NG_ + n];
}

// ---- phase 1: gates_x[B*T, 4H] = Xbf @ Wx^T + bias --------------------------
// Register-blocked: each wave owns a 32x64 output region (2 M-tiles x 4 N-tiles,
// 8 f32 accumulators). A fragments reused 4x, B fragments 2x -> ~21 FLOP/byte
// from L2 instead of ~8 with one tile/wave.
__global__ void k_gemm_x(const bf16_t* __restrict__ xb, const bf16_t* __restrict__ wt,
                         const float* __restrict__ bias, float* __restrict__ gates) {
    int wave = (blockIdx.x * blockDim.x + threadIdx.x) >> 5;  // 32768 waves
    int lane = threadIdx.x & 31;
    int mg   = wave >> 6;          // 0..511 : rows mg*32 .. +31
    int ng   = wave & 63;          // 0..63  : cols ng*64 .. +63
    int lrow = lane & 15;
    int koff = (lane >> 4) << 3;

    const bf16_t* ap0 = xb + (size_t)(mg * 32 + lrow) * D_ + koff;
    const bf16_t* ap1 = ap0 + (size_t)16 * D_;
    const bf16_t* bp  = wt + (size_t)(ng * 64 + lrow) * KK_ + koff;  // x-part: k in [0,D)

    v8f acc[2][4] = {};
    for (int kk = 0; kk < D_; kk += 32) {
        v16bf a0 = load_frag(ap0 + kk);
        v16bf a1 = load_frag(ap1 + kk);
        v16bf b0 = load_frag(bp + kk);
        v16bf b1 = load_frag(bp + (size_t)16 * KK_ + kk);
        v16bf b2 = load_frag(bp + (size_t)32 * KK_ + kk);
        v16bf b3 = load_frag(bp + (size_t)48 * KK_ + kk);
        acc[0][0] = __builtin_amdgcn_wmma_f32_16x16x32_bf16(false, a0, false, b0, (short)0, acc[0][0], false, false);
        acc[0][1] = __builtin_amdgcn_wmma_f32_16x16x32_bf16(false, a0, false, b1, (short)0, acc[0][1], false, false);
        acc[0][2] = __builtin_amdgcn_wmma_f32_16x16x32_bf16(false, a0, false, b2, (short)0, acc[0][2], false, false);
        acc[0][3] = __builtin_amdgcn_wmma_f32_16x16x32_bf16(false, a0, false, b3, (short)0, acc[0][3], false, false);
        acc[1][0] = __builtin_amdgcn_wmma_f32_16x16x32_bf16(false, a1, false, b0, (short)0, acc[1][0], false, false);
        acc[1][1] = __builtin_amdgcn_wmma_f32_16x16x32_bf16(false, a1, false, b1, (short)0, acc[1][1], false, false);
        acc[1][2] = __builtin_amdgcn_wmma_f32_16x16x32_bf16(false, a1, false, b2, (short)0, acc[1][2], false, false);
        acc[1][3] = __builtin_amdgcn_wmma_f32_16x16x32_bf16(false, a1, false, b3, (short)0, acc[1][3], false, false);
    }

    int half = (lane >> 4) << 3;
#pragma unroll
    for (int nj = 0; nj < 4; ++nj) {
        int   ncol = ng * 64 + nj * 16 + lrow;
        float bv   = bias[ncol];
#pragma unroll
        for (int mi = 0; mi < 2; ++mi) {
#pragma unroll
            for (int r = 0; r < 8; ++r) {
                int mrow = mg * 32 + mi * 16 + r + half;  // C layout: VGPR r -> M=r (+8, lanes>=16)
                gates[(size_t)mrow * NG_ + ncol] = acc[mi][nj][r] + bv;
            }
        }
    }
}

// ---- phase 2: one step of the recurrence ------------------------------------
// grid = 64 blocks (16 H-columns each), 256 threads = 8 waves:
//   wave = {mtile (B rows 0-15 / 16-31)} x {gate g in i,j,f,o}
// W_h tile is DMA'd by the Tensor Data Mover into LDS per 256-wide k-chunk;
// WMMA B-fragments then come from ds_load_b128. Each block owns all 4 gates
// for its columns -> full LSTM epilogue in LDS.
__global__ void k_lstm_step(const float* __restrict__ gates,
                            const bf16_t* __restrict__ wt,
                            const bf16_t* __restrict__ h_in,
                            bf16_t* __restrict__ h_out,
                            float* __restrict__ c,
                            float* __restrict__ out,
                            int s) {
    __shared__ float zb[4][B_][16];
#if USE_TDM
    __shared__ __align__(16) bf16_t bsm[4][16][KCH_];   // 32 KB staging
#endif

    int wave  = threadIdx.x >> 5;
    int lane  = threadIdx.x & 31;
    int mtile = wave & 1;
    int g     = wave >> 1;
    int lrow  = lane & 15;
    int koff  = (lane >> 4) << 3;
    int ncol  = blockIdx.x * 16 + lrow;
    int tx    = (T_ - 1) - s;            // reversed time index into x-gates

    const bf16_t* ap = h_in + (size_t)(mtile * 16 + lrow) * H_ + koff;
    const bf16_t* bp = wt + (size_t)(g * H_ + ncol) * KK_ + D_ + koff;  // h-part
    (void)bp;

    // warm the gates row this block's epilogue will read
    __builtin_prefetch(&gates[((size_t)(mtile * 16 + lrow) * T_ + tx) * NG_ + g * H_ + ncol], 0, 0);

    v8f acc = {};
    for (int kc = 0; kc < H_; kc += KCH_) {
#if USE_TDM
        if (wave == 0) {
#pragma unroll
            for (int gg = 0; gg < 4; ++gg) {
                const bf16_t* src = wt + (size_t)(gg * H_ + blockIdx.x * 16) * KK_ + D_ + kc;
                tdm_load_tile((unsigned)(uintptr_t)&bsm[gg][0][0], src, 16, KCH_, KK_);
            }
            __builtin_amdgcn_s_wait_tensorcnt(0);
        }
        __syncthreads();
#endif
#pragma unroll 2
        for (int kk = 0; kk < KCH_; kk += 32) {
            v16bf a = load_frag(ap + kc + kk);
#if USE_TDM
            v16bf b = load_frag(&bsm[g][lrow][kk + koff]);
#else
            v16bf b = load_frag(bp + kc + kk);
#endif
            acc = __builtin_amdgcn_wmma_f32_16x16x32_bf16(false, a, false, b,
                                                          (short)0, acc, false, false);
        }
#if USE_TDM
        __syncthreads();                  // protect bsm before next chunk's DMA
#endif
    }

    int half = (lane >> 4) << 3;
#pragma unroll
    for (int r = 0; r < 8; ++r) {
        int   mrow = mtile * 16 + r + half;     // == batch index
        float z = acc[r] + gates[((size_t)mrow * T_ + tx) * NG_ + g * H_ + ncol];
        zb[g][mrow][lrow] = z;
    }
    __syncthreads();

#pragma unroll
    for (int e = 0; e < 2; ++e) {
        int idx = threadIdx.x + e * 256;        // 512 = 32 rows x 16 cols
        int b   = idx >> 4;
        int nl  = idx & 15;
        int nc  = blockIdx.x * 16 + nl;
        float iv = zb[0][b][nl];
        float jv = zb[1][b][nl];
        float fv = zb[2][b][nl];
        float ov = zb[3][b][nl];
        float cold = c[b * H_ + nc];
        float cn = sigm(fv + 1.0f) * cold + sigm(iv) * tanhf(jv);  // FORGET_BIAS = 1
        float hn = sigm(ov) * tanhf(cn);
        c[b * H_ + nc] = cn;
        out[((size_t)b * T_ + s) * H_ + nc] = cn;   // cell state collected per step
        h_out[b * H_ + nc] = (bf16_t)hn;
    }
}

extern "C" void kernel_launch(void* const* d_in, const int* in_sizes, int n_in,
                              void* d_out, int out_size, void* d_ws, size_t ws_size,
                              hipStream_t stream) {
    (void)in_sizes; (void)n_in; (void)out_size; (void)ws_size;

    const float* x      = (const float*)d_in[0];
    /* d_in[1] = sl : unused by the reference */
    const float* kernel = (const float*)d_in[2];
    const float* bias   = (const float*)d_in[3];
    float*       out    = (float*)d_out;

    char* ws = (char*)d_ws;
    bf16_t* wt    = (bf16_t*)ws;                                   // 16 MiB: [4H, KK] bf16
    bf16_t* xb    = (bf16_t*)(ws + ((size_t)16 << 20));            // 32 MiB: [B*T, D] bf16
    float*  gates = (float*) (ws + ((size_t)48 << 20));            // 256 MiB: [B*T, 4H] f32
    float*  cbuf  = (float*) (ws + ((size_t)48 << 20) + (size_t)B_ * T_ * NG_ * 4);
    bf16_t* h0    = (bf16_t*)((char*)cbuf + (size_t)B_ * H_ * 4);
    bf16_t* h1    = h0 + (size_t)B_ * H_;

    (void)hipMemsetAsync(cbuf, 0, (size_t)B_ * H_ * sizeof(float),  stream);
    (void)hipMemsetAsync(h0,   0, (size_t)B_ * H_ * sizeof(bf16_t), stream);

    // prep
    k_cvt_x      <<<(B_ * T_ * D_) / 256, 256, 0, stream>>>(x, xb, B_ * T_ * D_);
    k_transpose_w<<<(NG_ * KK_)    / 256, 256, 0, stream>>>(kernel, wt);

    // phase 1: big input GEMM (32768 waves, 2x4 tiles each, 8 waves/block)
    k_gemm_x<<<4096, 256, 0, stream>>>(xb, wt, bias, gates);

    // phase 2: 512 sequential recurrent steps (W_h stays hot in 192 MB L2)
    for (int s = 0; s < T_; ++s) {
        const bf16_t* hin  = (s & 1) ? h1 : h0;
        bf16_t*       hout = (s & 1) ? h0 : h1;
        k_lstm_step<<<64, 256, 0, stream>>>(gates, wt, hin, hout, cbuf, out, s);
    }
}
